// FeatureExtractor_34333968564270
// MI455X (gfx1250) — matrix operand order repaired and checked by reading, not compiled, not run on hardware
//
#include <hip/hip_runtime.h>

#define S_LEN 512
#define D_DIM 64
#define KSTR 72      // f16 elems per K row  (bank-rotating, 16B-aligned rows)
#define VSTR 520     // f16 elems per Vt row
#define WSTR 72      // f16 elems per weight row
#define LN_EPS 1e-5f

typedef _Float16 f16;
typedef __attribute__((ext_vector_type(16))) _Float16 v16h;
typedef __attribute__((ext_vector_type(8)))  _Float16 v8h;
typedef __attribute__((ext_vector_type(2)))  _Float16 v2h;
typedef __attribute__((ext_vector_type(8)))  float    v8f;

struct __align__(16) Smem {
  f16 K[S_LEN * KSTR];            // keys  [s][d], f16, padded          73,728 B
  f16 Vt[D_DIM * VSTR];           // values transposed [d][s], f16      66,560 B
  union {
    f16   w0[D_DIM * WSTR];       // phase A: Wk ; phase B: Wq           9,216 B
    float red[8][D_DIM];          // final cross-wave reduction (after barrier)
  };
  f16   w1[D_DIM * WSTR];         // phase A: Wv ; phase B: Wo           9,216 B
  float x[S_LEN];                 // phase A: x_kv ; phase B: x_q        2,048 B
  float pw[D_DIM], pb[D_DIM];     //                                       512 B
  float vA[D_DIM], vB[D_DIM], vC[D_DIM], vD[D_DIM]; // phase vectors     1,024 B
};                                // total: 162,304 B

__device__ __forceinline__ v8f wmma_f16(v16h a, v16h b, v8f c) {
  return __builtin_amdgcn_wmma_f32_16x16x32_f16(false, a, false, b, (short)0, c,
                                                false, false);
}

// A-operand (16 rows x 32 cols, f16) from row-major LDS tile.
// lane: m = lane&15, elems e: col = (lane>>4)*16 + e  (16 contiguous f16)
__device__ __forceinline__ v16h load_a(const f16* p00, int stride, int lane) {
  const int m = lane & 15, h = lane >> 4;
  const f16* p = p00 + m * stride + h * 16;
  v8h lo = *(const v8h*)p;
  v8h hi = *(const v8h*)(p + 8);
  v16h r;
#pragma unroll
  for (int i = 0; i < 8; ++i) { r[i] = lo[i]; r[i + 8] = hi[i]; }
  return r;
}

// B-operand (32 k x 16 n, f16) where LDS holds B^T[n][k] row-major.
// lane: n = lane&15, elems e: k = (e>>3)*16 + (lane>>4)*8 + (e&7)
__device__ __forceinline__ v16h load_bt(const f16* p00, int stride, int lane) {
  const int n = lane & 15, h = lane >> 4;
  const f16* p = p00 + n * stride + h * 8;
  v8h lo = *(const v8h*)p;
  v8h hi = *(const v8h*)(p + 16);
  v16h r;
#pragma unroll
  for (int i = 0; i < 8; ++i) { r[i] = lo[i]; r[i + 8] = hi[i]; }
  return r;
}

// On-the-fly h = x*pw+pb as a B operand (h^T[n=s][k=d]) in f16.
__device__ __forceinline__ v16h make_h_b(const Smem& sm, int sbase, int dbase,
                                         int lane) {
  const int n = lane & 15, h = lane >> 4;
  const float xs = sm.x[sbase + n];
  const int d0 = dbase + h * 16;
  v16h r;
#pragma unroll
  for (int e = 0; e < 16; ++e)
    r[e] = (f16)(xs * sm.pw[d0 + e] + sm.pb[d0 + e]);
  return r;
}

__device__ __forceinline__ float xor16(float v) { return __shfl_xor(v, 16, 32); }

__global__ __launch_bounds__(256)
void fused_xattn_kernel(const float* __restrict__ x_src, const float* __restrict__ x_tgt,
                        const float* __restrict__ proj_w, const float* __restrict__ proj_b,
                        const float* __restrict__ sa_in_w, const float* __restrict__ sa_in_b,
                        const float* __restrict__ sa_out_w, const float* __restrict__ sa_out_b,
                        const float* __restrict__ sa_ln_g, const float* __restrict__ sa_ln_b,
                        const float* __restrict__ ca_in_w, const float* __restrict__ ca_in_b,
                        const float* __restrict__ ca_out_w, const float* __restrict__ ca_out_b,
                        const float* __restrict__ ca_ln_g, const float* __restrict__ ca_ln_b,
                        float* __restrict__ out, int Bn) {
  __shared__ Smem sm;
  const int b    = blockIdx.x;
  const int task = blockIdx.y;           // 0: self-src, 1: self-tgt, 2: cross
  const int tid  = threadIdx.x;
  const int lane = tid & 31, wave = tid >> 5;
  const int h    = lane >> 4;
  const bool hih = (h == 1);

  const float* in_w  = (task == 2) ? ca_in_w  : sa_in_w;
  const float* in_b  = (task == 2) ? ca_in_b  : sa_in_b;
  const float* out_w = (task == 2) ? ca_out_w : sa_out_w;
  const float* out_b = (task == 2) ? ca_out_b : sa_out_b;
  const float* ln_g  = (task == 2) ? ca_ln_g  : sa_ln_g;
  const float* ln_b  = (task == 2) ? ca_ln_b  : sa_ln_b;
  const float* xq_g  = (task == 1) ? x_tgt : x_src;
  const float* xkv_g = (task == 0) ? x_src : x_tgt;

  // ---------------- phase 0: load x_kv, proj, Wk/Wv, bk/bv ----------------
  for (int i = tid; i < S_LEN; i += 256) sm.x[i] = xkv_g[(size_t)b * S_LEN + i];
  if (tid < D_DIM) {
    sm.pw[tid] = proj_w[tid];
    sm.pb[tid] = proj_b[tid];
    sm.vA[tid] = in_b[D_DIM + tid];       // bk
    sm.vB[tid] = in_b[2 * D_DIM + tid];   // bv
  }
  for (int i = tid; i < D_DIM * D_DIM; i += 256) {
    int r = i >> 6, c = i & 63;
    sm.w0[r * WSTR + c] = (f16)in_w[(D_DIM + r) * D_DIM + c];      // Wk
    sm.w1[r * WSTR + c] = (f16)in_w[(2 * D_DIM + r) * D_DIM + c];  // Wv
  }
  __syncthreads();

  // ---------------- phase A: K = h_kv Wk^T + bk ; Vt = (h_kv Wv^T + bv)^T --
#pragma unroll 1
  for (int i = 0; i < 4; ++i) {
    const int st = wave * 4 + i;
    const int sbase = st * 16;
    const int s_lane = sbase + (lane & 15);
    v16h hb0 = make_h_b(sm, sbase, 0, lane);
    v16h hb1 = make_h_b(sm, sbase, 32, lane);
#pragma unroll
    for (int t = 0; t < 4; ++t) {
      v8f ck = {};
      ck = wmma_f16(load_a(&sm.w0[(t * 16) * WSTR + 0], WSTR, lane), hb0, ck);
      ck = wmma_f16(load_a(&sm.w0[(t * 16) * WSTR + 32], WSTR, lane), hb1, ck);
#pragma unroll
      for (int j = 0; j < 4; ++j) {        // K^T tile -> K row-major (paired)
        int dk = t * 16 + 2 * j + 8 * h;
        v2h pr;
        pr[0] = (f16)(ck[2 * j]     + sm.vA[dk]);
        pr[1] = (f16)(ck[2 * j + 1] + sm.vA[dk + 1]);
        *(v2h*)&sm.K[s_lane * KSTR + dk] = pr;
      }
      v8f cv = {};
      cv = wmma_f16(load_a(&sm.w1[(t * 16) * WSTR + 0], WSTR, lane), hb0, cv);
      cv = wmma_f16(load_a(&sm.w1[(t * 16) * WSTR + 32], WSTR, lane), hb1, cv);
#pragma unroll
      for (int r = 0; r < 8; ++r) {        // V^T tile -> Vt row-major
        int dv = t * 16 + r + 8 * h;
        sm.Vt[dv * VSTR + s_lane] = (f16)(cv[r] + sm.vB[dv]);
      }
    }
  }
  __syncthreads();

  // ---------------- phase B loads: x_q, Wq/Wo, bq/bo/ln -------------------
  for (int i = tid; i < S_LEN; i += 256) sm.x[i] = xq_g[(size_t)b * S_LEN + i];
  if (tid < D_DIM) {
    sm.vA[tid] = in_b[tid];    // bq
    sm.vB[tid] = out_b[tid];   // bo
    sm.vC[tid] = ln_g[tid];
    sm.vD[tid] = ln_b[tid];
  }
  for (int i = tid; i < D_DIM * D_DIM; i += 256) {
    int r = i >> 6, c = i & 63;
    sm.w0[r * WSTR + c] = (f16)in_w[r * D_DIM + c];    // Wq
    sm.w1[r * WSTR + c] = (f16)out_w[r * D_DIM + c];   // Wo
  }
  __syncthreads();

  // ---------------- phase C: flash attention per 16-query tile ------------
  float acc[4] = {0.f, 0.f, 0.f, 0.f};  // per-lane z sums per 16-wide d-tile
#pragma unroll 1
  for (int qi = 0; qi < 4; ++qi) {
    const int qbase = (wave * 4 + qi) * 16;
    // Q^T = Wq h_q^T + bq (scaled by 1/sqrt(D))
    v16h hb0 = make_h_b(sm, qbase, 0, lane);
    v16h hb1 = make_h_b(sm, qbase, 32, lane);
    float qt[4][8];
#pragma unroll
    for (int t = 0; t < 4; ++t) {
      v8f c = {};
      c = wmma_f16(load_a(&sm.w0[(t * 16) * WSTR + 0], WSTR, lane), hb0, c);
      c = wmma_f16(load_a(&sm.w0[(t * 16) * WSTR + 32], WSTR, lane), hb1, c);
#pragma unroll
      for (int r = 0; r < 8; ++r)
        qt[t][r] = (c[r] + sm.vA[t * 16 + r + 8 * h]) * 0.125f;
    }
    // exchange row-halves (lane <-> lane^16) and build B operands of Q
    v16h qB0, qB1;
#pragma unroll
    for (int r = 0; r < 8; ++r) {
      float e0 = xor16(qt[0][r]), e1 = xor16(qt[1][r]);
      float e2 = xor16(qt[2][r]), e3 = xor16(qt[3][r]);
      qB0[r]     = (f16)(hih ? e1 : qt[0][r]);
      qB0[r + 8] = (f16)(hih ? qt[1][r] : e0);
      qB1[r]     = (f16)(hih ? e3 : qt[2][r]);
      qB1[r + 8] = (f16)(hih ? qt[3][r] : e2);
    }

    float mrun = -3.0e38f, lrun = 0.f;
    v8f o0 = {}, o1 = {}, o2 = {}, o3 = {};
#pragma unroll 1
    for (int c = 0; c < 16; ++c) {         // 32 keys per chunk
      const int kb = c * 32;
      v8f sA = {}, sB = {};                // S^T = K Q^T
      sA = wmma_f16(load_a(&sm.K[(kb)*KSTR + 0], KSTR, lane), qB0, sA);
      sA = wmma_f16(load_a(&sm.K[(kb)*KSTR + 32], KSTR, lane), qB1, sA);
      sB = wmma_f16(load_a(&sm.K[(kb + 16) * KSTR + 0], KSTR, lane), qB0, sB);
      sB = wmma_f16(load_a(&sm.K[(kb + 16) * KSTR + 32], KSTR, lane), qB1, sB);
      float cm = sA[0];
#pragma unroll
      for (int r = 0; r < 8; ++r) { cm = fmaxf(cm, sA[r]); cm = fmaxf(cm, sB[r]); }
      cm = fmaxf(cm, xor16(cm));
      const float nm = fmaxf(mrun, cm);
      const float scale = __expf(mrun - nm);
      mrun = nm;
      float pA[8], pB[8], csum = 0.f;
#pragma unroll
      for (int r = 0; r < 8; ++r) {
        pA[r] = __expf(sA[r] - nm);
        pB[r] = __expf(sB[r] - nm);
        csum += pA[r] + pB[r];
      }
      csum += xor16(csum);
      lrun = lrun * scale + csum;
#pragma unroll
      for (int r = 0; r < 8; ++r) {
        o0[r] *= scale; o1[r] *= scale; o2[r] *= scale; o3[r] *= scale;
      }
      v16h P;                              // probabilities as B operand
#pragma unroll
      for (int r = 0; r < 8; ++r) {
        float eA = xor16(pA[r]), eB = xor16(pB[r]);
        P[r]     = (f16)(hih ? eB : pA[r]);
        P[r + 8] = (f16)(hih ? pB[r] : eA);
      }
      o0 = wmma_f16(load_a(&sm.Vt[0 * VSTR + kb], VSTR, lane), P, o0);
      o1 = wmma_f16(load_a(&sm.Vt[16 * VSTR + kb], VSTR, lane), P, o1);
      o2 = wmma_f16(load_a(&sm.Vt[32 * VSTR + kb], VSTR, lane), P, o2);
      o3 = wmma_f16(load_a(&sm.Vt[48 * VSTR + kb], VSTR, lane), P, o3);
    }

    // finalize O (divide by l), re-interpret O^T regs as A operand of O
    const float linv = 1.0f / lrun;
    v16h A0, A1;
#pragma unroll
    for (int r = 0; r < 8; ++r) {
      A0[r]     = (f16)(o0[r] * linv);
      A0[r + 8] = (f16)(o1[r] * linv);
      A1[r]     = (f16)(o2[r] * linv);
      A1[r + 8] = (f16)(o3[r] * linv);
    }
    // Y = O Wo^T
    v8f y[4];
#pragma unroll
    for (int nt = 0; nt < 4; ++nt) {
      v8f c = {};
      c = wmma_f16(A0, load_bt(&sm.w1[(nt * 16) * WSTR + 0], WSTR, lane), c);
      c = wmma_f16(A1, load_bt(&sm.w1[(nt * 16) * WSTR + 32], WSTR, lane), c);
      y[nt] = c;
    }
    // + bo + residual h_q (exact f32 recompute)
    float xr[8];
#pragma unroll
    for (int r = 0; r < 8; ++r) xr[r] = sm.x[qbase + r + 8 * h];
#pragma unroll
    for (int nt = 0; nt < 4; ++nt) {
      const int dd = nt * 16 + (lane & 15);
      const float pwv = sm.pw[dd], pbv = sm.pb[dd], bov = sm.vB[dd];
#pragma unroll
      for (int r = 0; r < 8; ++r) y[nt][r] += bov + xr[r] * pwv + pbv;
    }
    // layernorm over d=64 per row q (16-lane xor reductions)
    float mu[8], inv[8];
#pragma unroll
    for (int r = 0; r < 8; ++r) {
      float s = y[0][r] + y[1][r] + y[2][r] + y[3][r];
      s += __shfl_xor(s, 1, 32); s += __shfl_xor(s, 2, 32);
      s += __shfl_xor(s, 4, 32); s += __shfl_xor(s, 8, 32);
      mu[r] = s * (1.0f / 64.0f);
    }
#pragma unroll
    for (int r = 0; r < 8; ++r) {
      float d0 = y[0][r] - mu[r], d1 = y[1][r] - mu[r];
      float d2 = y[2][r] - mu[r], d3 = y[3][r] - mu[r];
      float s = d0 * d0 + d1 * d1 + d2 * d2 + d3 * d3;
      s += __shfl_xor(s, 1, 32); s += __shfl_xor(s, 2, 32);
      s += __shfl_xor(s, 4, 32); s += __shfl_xor(s, 8, 32);
      inv[r] = rsqrtf(s * (1.0f / 64.0f) + LN_EPS);
    }
#pragma unroll
    for (int nt = 0; nt < 4; ++nt) {
      const int dd = nt * 16 + (lane & 15);
      const float g = sm.vC[dd], bb = sm.vD[dd];
      float al = 0.f;
#pragma unroll
      for (int r = 0; r < 8; ++r)
        al += (y[nt][r] - mu[r]) * inv[r] * g + bb;
      acc[nt] += al;
    }
  }

  // ---------------- mean over S: cross-half + cross-wave reduction --------
#pragma unroll
  for (int nt = 0; nt < 4; ++nt) acc[nt] += xor16(acc[nt]);
  __syncthreads();                 // all waves done with w0 -> alias as red
  if (lane < 16) {
#pragma unroll
    for (int nt = 0; nt < 4; ++nt) sm.red[wave][nt * 16 + lane] = acc[nt];
  }
  __syncthreads();
  if (tid < D_DIM) {
    float s = 0.f;
#pragma unroll
    for (int w = 0; w < 8; ++w) s += sm.red[w][tid];
    const size_t BD = (size_t)Bn * D_DIM;
    const size_t off = (task == 2) ? 0 : ((task == 0) ? BD : 2 * BD);
    out[off + (size_t)b * D_DIM + tid] = s * (1.0f / (float)S_LEN);
  }
}

extern "C" void kernel_launch(void* const* d_in, const int* in_sizes, int n_in,
                              void* d_out, int out_size, void* d_ws, size_t ws_size,
                              hipStream_t stream) {
  (void)n_in; (void)out_size; (void)d_ws; (void)ws_size;
  const float* x_src    = (const float*)d_in[0];
  const float* x_tgt    = (const float*)d_in[1];
  const float* proj_w   = (const float*)d_in[2];
  const float* proj_b   = (const float*)d_in[3];
  const float* sa_in_w  = (const float*)d_in[4];
  const float* sa_in_b  = (const float*)d_in[5];
  const float* sa_out_w = (const float*)d_in[6];
  const float* sa_out_b = (const float*)d_in[7];
  const float* sa_ln_g  = (const float*)d_in[8];
  const float* sa_ln_b  = (const float*)d_in[9];
  const float* ca_in_w  = (const float*)d_in[10];
  const float* ca_in_b  = (const float*)d_in[11];
  const float* ca_out_w = (const float*)d_in[12];
  const float* ca_out_b = (const float*)d_in[13];
  const float* ca_ln_g  = (const float*)d_in[14];
  const float* ca_ln_b  = (const float*)d_in[15];

  const int Bn = in_sizes[0] / S_LEN;
  dim3 grid(Bn, 3), block(256);
  fused_xattn_kernel<<<grid, block, 0, stream>>>(
      x_src, x_tgt, proj_w, proj_b, sa_in_w, sa_in_b, sa_out_w, sa_out_b,
      sa_ln_g, sa_ln_b, ca_in_w, ca_in_b, ca_out_w, ca_out_b, ca_ln_g, ca_ln_b,
      (float*)d_out, Bn);
}
